// AttentionWithContextV2_58334245814317
// MI455X (gfx1250) — compile-verified, matrix operand order checked by reading, not run on hardware
//
#include <hip/hip_runtime.h>
#include <hip/hip_bf16.h>
#include <math.h>

typedef __attribute__((ext_vector_type(16))) __bf16 v16bf;
typedef __attribute__((ext_vector_type(8)))  float  v8f;

constexpr int Bb = 64, Tt = 2048, Ff = 256, Cc = 64;
constexpr float EPSc = 1e-7f;

// workspace layout (bytes)
constexpr size_t WT_OFF  = 0;                              // bf16 Wt[n][k] = W[k][n]
constexpr size_t UB_OFF  = WT_OFF + (size_t)Ff * Ff * 2;   // bf16 ub[c][k]
constexpr size_t E_OFF   = 163840;                         // f32 e[B][T][C]
constexpr size_t DEN_OFF = E_OFF + (size_t)Bb * Tt * Cc * 4; // f32 denom[B][C]

__device__ __forceinline__ unsigned short f2bf(float f) {
  __hip_bfloat16 h = __float2bfloat16(f);
  return *reinterpret_cast<unsigned short*>(&h);
}

__device__ __forceinline__ v8f zero8() {
  v8f z = {0.f, 0.f, 0.f, 0.f, 0.f, 0.f, 0.f, 0.f};
  return z;
}

// Gather a 16x32 bf16 WMMA fragment from a row-major [row][k] bf16 array.
// Layout per ISA 7.12.2: lane<16 -> K 0..7 (vgpr0-3) + 16..23 (vgpr4-7);
// lane>=16 -> K 8..15 + 24..31. Works for A (rows=M) and B (rows=N) frags.
__device__ __forceinline__ v16bf load_frag16(const unsigned short* p, int strideHalfs,
                                             int rowBase, int kBase, int lane) {
  const int r  = rowBase + (lane & 15);
  const int kb = kBase + ((lane >> 4) << 3);
  const unsigned short* q = p + (size_t)r * strideHalfs + kb;
  union { uint4 u[2]; v16bf v; } f;
  f.u[0] = *reinterpret_cast<const uint4*>(q);
  f.u[1] = *reinterpret_cast<const uint4*>(q + 16);
  return f.v;
}

__device__ __forceinline__ v8f wmma_bf16(v16bf a, v16bf b, v8f c) {
  return __builtin_amdgcn_wmma_f32_16x16x32_bf16(false, a, false, b, (short)0, c,
                                                 false, false);
}

__device__ __forceinline__ float fast_tanh(float x) {
  x = fminf(fmaxf(x, -15.f), 15.f);
  float e2 = __expf(2.f * x);
  return (e2 - 1.f) / (e2 + 1.f);
}

// ---------------------------------------------------------------------------
// prep: zero output-gemm region + denom; convert W (transposed) and u to bf16
// ---------------------------------------------------------------------------
__global__ __launch_bounds__(256) void prep_kernel(
    const float* __restrict__ W, const float* __restrict__ u,
    unsigned short* __restrict__ Wt, unsigned short* __restrict__ ub,
    float* __restrict__ denom, float* __restrict__ out_zero) {
  const int N0 = Bb * Cc * Ff;    // zero output
  const int N1 = N0 + Ff * Ff;    // Wt
  const int N2 = N1 + Cc * Ff;    // ub
  const int N3 = N2 + Bb * Cc;    // zero denom
  int idx = blockIdx.x * blockDim.x + threadIdx.x;
  if (idx < N0) {
    out_zero[idx] = 0.f;
  } else if (idx < N1) {
    int t = idx - N0;
    int k = t & (Ff - 1), n = t >> 8;            // Wt[n][k] = W[k][n]
    Wt[t] = f2bf(W[(size_t)k * Ff + n]);
  } else if (idx < N2) {
    int t = idx - N1;
    ub[t] = f2bf(u[t]);
  } else if (idx < N3) {
    denom[idx - N2] = 0.f;
  }
}

// ---------------------------------------------------------------------------
// k1: per 64-row tile:  uit = tanh(xW + b);  e = exp(uit uT) * mask
//     also accumulates denom[b][c] = sum_t e
// Wave assignment (GEMM1): each wave owns N-tiles and iterates ALL M-tiles, so each
// global Wt fragment is fetched exactly once per block (ub frags twice, GEMM2).
// ---------------------------------------------------------------------------
__global__ __launch_bounds__(256) void k1_tanh_exp(
    const float* __restrict__ x, const float* __restrict__ bias,
    const unsigned char* __restrict__ mask,
    const unsigned short* __restrict__ Wt, const unsigned short* __restrict__ ub,
    float* __restrict__ e, float* __restrict__ denom) {
  constexpr int TM = 64;
  constexpr int LS = Ff + 8;  // 264 halfs: padded to avoid bank conflicts
  __shared__ __align__(16) unsigned short sA[TM * LS];  // x tile, then uit tile
  __shared__ float smask[TM];

  const int tid = threadIdx.x;
  const int b   = blockIdx.x / (Tt / TM);
  const int t0  = (blockIdx.x % (Tt / TM)) * TM;

  for (int idx = tid; idx < TM * Ff; idx += 256) {
    int r = idx >> 8, f = idx & (Ff - 1);
    sA[r * LS + f] = f2bf(x[((size_t)b * Tt + t0 + r) * Ff + f]);
  }
  if (tid < TM) smask[tid] = mask[(size_t)b * Tt + t0 + tid] ? 1.f : 0.f;
  __syncthreads();

  const int lane = tid & 31;
  const int w    = tid >> 5;

  // ---- GEMM1: [64 x 256] = sA * Wt^T. Wave w owns N-tiles {2w, 2w+1},
  //      iterates M-tiles 0..3 (A frags from LDS, hoisted per k-step). ----
  v8f acc[2][4];
#pragma unroll
  for (int jn = 0; jn < 2; ++jn)
#pragma unroll
    for (int mi = 0; mi < 4; ++mi) acc[jn][mi] = zero8();

  for (int k0 = 0; k0 < Ff; k0 += 32) {
    v16bf va[4];
#pragma unroll
    for (int mi = 0; mi < 4; ++mi) va[mi] = load_frag16(sA, LS, mi * 16, k0, lane);
#pragma unroll
    for (int jn = 0; jn < 2; ++jn) {
      v16bf bfr = load_frag16(Wt, Ff, (2 * w + jn) * 16, k0, lane);
#pragma unroll
      for (int mi = 0; mi < 4; ++mi)
        acc[jn][mi] = wmma_bf16(va[mi], bfr, acc[jn][mi]);
    }
  }

  // bias + tanh, write uit back into sA (bf16)
  __syncthreads();
#pragma unroll
  for (int jn = 0; jn < 2; ++jn) {
    int ncol = (2 * w + jn) * 16 + (lane & 15);
    float bv = bias[ncol];
#pragma unroll
    for (int mi = 0; mi < 4; ++mi) {
#pragma unroll
      for (int i = 0; i < 8; ++i) {
        int m = mi * 16 + ((lane >> 4) << 3) + i;
        sA[m * LS + ncol] = f2bf(fast_tanh(acc[jn][mi][i] + bv));
      }
    }
  }
  __syncthreads();

  // ---- GEMM2: ait[64 x 64] = uit * u^T. Wave w owns N-tile (w&3),
  //      M-tiles {2*(w>>2), 2*(w>>2)+1}. ub frags fetched twice per block. ----
  const int nt2  = (w & 3) * 16;
  const int mg2  = (w >> 2) * 2;
  v8f acc2[2];
  acc2[0] = zero8();
  acc2[1] = zero8();
  for (int k0 = 0; k0 < Ff; k0 += 32) {
    v16bf bfr = load_frag16(ub, Ff, nt2, k0, lane);
#pragma unroll
    for (int mi = 0; mi < 2; ++mi) {
      v16bf a = load_frag16(sA, LS, (mg2 + mi) * 16, k0, lane);
      acc2[mi] = wmma_bf16(a, bfr, acc2[mi]);
    }
  }

  // exp * mask -> e, plus per-column partial sums into denom
  const int c = nt2 + (lane & 15);
#pragma unroll
  for (int mi = 0; mi < 2; ++mi) {
    float s = 0.f;
#pragma unroll
    for (int i = 0; i < 8; ++i) {
      int m   = (mg2 + mi) * 16 + ((lane >> 4) << 3) + i;
      float v = __expf(acc2[mi][i]) * smask[m];
      e[((size_t)b * Tt + t0 + m) * Cc + c] = v;
      s += v;
    }
    s += __shfl_xor(s, 16, 32);
    if (lane < 16) atomicAdd(&denom[b * Cc + c], s);
  }
}

// ---------------------------------------------------------------------------
// k2: a = e/(denom+eps); scores[b][c][t] = a; output[b][c][f] += a^T * x
//     grid = B x 8 K-splits (each block owns 256 t's)
// ---------------------------------------------------------------------------
__global__ __launch_bounds__(256) void k2_out(
    const float* __restrict__ x, const float* __restrict__ e,
    const float* __restrict__ denom,
    float* __restrict__ outp, float* __restrict__ scores) {
  constexpr int KS = Tt / 8;    // 256 timesteps per block
  constexpr int LT = 32 + 8;    // padded stride (halfs)
  __shared__ __align__(16) unsigned short sAT[Cc * LT];  // a^T [c][t] bf16
  __shared__ __align__(16) unsigned short sXT[Ff * LT];  // x^T [f][t] bf16
  __shared__ float sAF[Cc * 33];                         // a   [c][t] f32 (exact)
  __shared__ float srden[Cc];

  const int tid = threadIdx.x;
  const int b   = blockIdx.x >> 3;
  const int ks  = blockIdx.x & 7;
  if (tid < Cc) srden[tid] = 1.f / (denom[b * Cc + tid] + EPSc);

  const int lane = tid & 31;
  const int w    = tid >> 5;
  const int mrow = (w & 3) * 16;  // C tile
  const int ng   = w >> 2;        // F half

  v8f acc[8];
#pragma unroll
  for (int j = 0; j < 8; ++j) acc[j] = zero8();

  for (int kk = 0; kk < 8; ++kk) {
    const int t0 = ks * KS + kk * 32;
    __syncthreads();  // previous-iteration reads done (also covers srden)
    // stage a^T (normalized), keep f32 copy for exact score stores
    for (int idx = tid; idx < 32 * Cc; idx += 256) {
      int tt = idx >> 6, c = idx & 63;
      float av = e[((size_t)b * Tt + t0 + tt) * Cc + c] * srden[c];
      sAT[c * LT + tt] = f2bf(av);
      sAF[c * 33 + tt] = av;
    }
    // stage x^T as bf16
    for (int idx = tid; idx < 32 * Ff; idx += 256) {
      int tt = idx >> 8, f = idx & (Ff - 1);
      sXT[f * LT + tt] = f2bf(x[((size_t)b * Tt + t0 + tt) * Ff + f]);
    }
    // prefetch next 32-t slab of x and e while this step computes
    if (kk < 7) {
      const float* nx = &x[((size_t)b * Tt + t0 + 32) * Ff];
      __builtin_prefetch(nx + (size_t)tid * 32, 0, 0);   // 256 thr * 128B = tile
      if (tid < 64) {
        const float* ne = &e[((size_t)b * Tt + t0 + 32) * Cc];
        __builtin_prefetch(ne + (size_t)tid * 32, 0, 0); // 64 thr * 128B = tile
      }
    }
    __syncthreads();
    // coalesced attention_scores stores (t contiguous per c)
    for (int idx = tid; idx < Cc * 32; idx += 256) {
      int c = idx >> 5, tt = idx & 31;
      scores[((size_t)b * Cc + c) * Tt + t0 + tt] = sAF[c * 33 + tt];
    }
    // GEMM3 step
    v16bf a = load_frag16(sAT, LT, mrow, 0, lane);
#pragma unroll
    for (int j = 0; j < 8; ++j) {
      v16bf bfr = load_frag16(sXT, LT, (ng * 8 + j) * 16, 0, lane);
      acc[j] = wmma_bf16(a, bfr, acc[j]);
    }
  }

  // accumulate K-split partials into output
#pragma unroll
  for (int j = 0; j < 8; ++j) {
    int f = (ng * 8 + j) * 16 + (lane & 15);
#pragma unroll
    for (int i = 0; i < 8; ++i) {
      int c = mrow + ((lane >> 4) << 3) + i;
      atomicAdd(&outp[((size_t)b * Cc + c) * Ff + f], acc[j][i]);
    }
  }
}

// ---------------------------------------------------------------------------
extern "C" void kernel_launch(void* const* d_in, const int* in_sizes, int n_in,
                              void* d_out, int out_size, void* d_ws, size_t ws_size,
                              hipStream_t stream) {
  const float* x          = (const float*)d_in[0];
  const float* W          = (const float*)d_in[1];
  const float* bias       = (const float*)d_in[2];
  const float* u          = (const float*)d_in[3];
  const unsigned char* mk = (const unsigned char*)d_in[4];

  float* outp   = (float*)d_out;                      // [B,C,F]
  float* scores = outp + (size_t)Bb * Cc * Ff;        // [B,C,T]

  char* ws = (char*)d_ws;
  unsigned short* Wt = (unsigned short*)(ws + WT_OFF);
  unsigned short* ub = (unsigned short*)(ws + UB_OFF);
  float* e           = (float*)(ws + E_OFF);
  float* denom       = (float*)(ws + DEN_OFF);

  const int totalPrep = Bb * Cc * Ff + Ff * Ff + Cc * Ff + Bb * Cc;
  prep_kernel<<<(totalPrep + 255) / 256, 256, 0, stream>>>(W, u, Wt, ub, denom, outp);
  k1_tanh_exp<<<Bb * (Tt / 64), 256, 0, stream>>>(x, bias, mk, Wt, ub, e, denom);
  k2_out<<<Bb * 8, 256, 0, stream>>>(x, e, denom, outp, scores);
}
